// customVGG_71588514890045
// MI455X (gfx1250) — matrix operand order repaired and checked by reading, not compiled, not run on hardware
//
#include <hip/hip_runtime.h>
#include <hip/hip_bf16.h>
#include <math.h>

typedef __attribute__((ext_vector_type(16))) _Float16 v16h;
typedef __attribute__((ext_vector_type(8)))  _Float16 v8h;
typedef __attribute__((ext_vector_type(8)))  float    v8f;

#define BN_EPS 1e-5f
#define BATCH  32
#define C1     64
#define C2     128
#define NNBR   20000
#define DFEAT  8192
#define K2     576      // 64 * 3 * 3

// workspace layout (float offsets)
#define WS_H1    0                         // 32*64*16*16  = 524288 floats
#define WS_Y2    (524288)                  // 32*128*16*16 = 1048576 floats
#define WS_NR    (WS_Y2 + 1048576)         // 32
#define WS_DR    (WS_NR + 32)              // 32
#define WS_PRED  (WS_DR + 32)              // 1 (int), +pad
#define WS_PACKA (WS_PRED + 8)             // 73728 halves = 36864 floats (32B aligned)

// ---------------------------------------------------------------------------
// Kernel 1: zero accumulators, bincount labels, argmax (smallest index on tie)
// ---------------------------------------------------------------------------
__global__ void init_stats_kernel(const int* __restrict__ labels, float* __restrict__ ws) {
    __shared__ int cnt[10];
    int tid = threadIdx.x;
    if (tid < 10) cnt[tid] = 0;
    if (tid < 32) { ws[WS_NR + tid] = 0.f; ws[WS_DR + tid] = 0.f; }
    __syncthreads();
    for (int i = tid; i < NNBR; i += blockDim.x)
        atomicAdd(&cnt[labels[i]], 1);
    __syncthreads();
    if (tid == 0) {
        int best = 0, bc = cnt[0];
        for (int c = 1; c < 10; ++c)
            if (cnt[c] > bc) { bc = cnt[c]; best = c; }   // strict > => smallest on tie
        ((int*)(ws + WS_PRED))[0] = best;
    }
}

// ---------------------------------------------------------------------------
// Kernel 2: repack conv2 weights into WMMA-A-fragment-linear f16.
// packA[((mt*18 + kt)*32 + lane)*16 + i] = w[oc=mt*16+l15][ic*9 + kk]
//   kt = kk*2 + ich, ic = ich*32 + 16*(i>>3) + 8*half + (i&7)
// ---------------------------------------------------------------------------
__global__ void pack_w2_kernel(const float* __restrict__ w, _Float16* __restrict__ packA) {
    int idx = blockIdx.x * blockDim.x + threadIdx.x;     // 8*18*32*16 = 73728
    if (idx >= 8 * 18 * 32 * 16) return;
    int i    = idx & 15;
    int lane = (idx >> 4) & 31;
    int kt   = (idx >> 9) % 18;
    int mt   = (idx >> 9) / 18;
    int half = lane >> 4, l15 = lane & 15;
    int kk = kt >> 1, ich = kt & 1;
    int oc = mt * 16 + l15;
    int ic = ich * 32 + ((i >> 3) << 4) + (half << 3) + (i & 7);
    packA[idx] = (_Float16)w[oc * K2 + ic * 9 + kk];
}

// ---------------------------------------------------------------------------
// Kernel 3: conv1(3x3,pad1) + BN + ReLU + 2x2 maxpool  -> h1 [32,64,16,16]
// ---------------------------------------------------------------------------
__global__ void conv1_kernel(const float* __restrict__ x, const float* __restrict__ w,
                             const float* __restrict__ bias,
                             const float* __restrict__ g, const float* __restrict__ beta,
                             const float* __restrict__ mean, const float* __restrict__ var,
                             float* __restrict__ h1) {
    int t = blockIdx.x * blockDim.x + threadIdx.x;       // 524288
    if (t >= BATCH * C1 * 16 * 16) return;
    int px = t & 15, py = (t >> 4) & 15, oc = (t >> 8) & 63, b = t >> 14;
    float sc = g[oc] * rsqrtf(var[oc] + BN_EPS);
    float sh = beta[oc] - mean[oc] * sc;
    float bi = bias[oc];
    float m = -INFINITY;
    #pragma unroll
    for (int dy = 0; dy < 2; ++dy)
    #pragma unroll
    for (int dx = 0; dx < 2; ++dx) {
        int oy = py * 2 + dy, ox = px * 2 + dx;
        float acc = bi;
        for (int ic = 0; ic < 3; ++ic)
            #pragma unroll
            for (int ky = 0; ky < 3; ++ky) {
                int iy = oy + ky - 1;
                if ((unsigned)iy >= 32u) continue;
                #pragma unroll
                for (int kx = 0; kx < 3; ++kx) {
                    int ix = ox + kx - 1;
                    if ((unsigned)ix >= 32u) continue;
                    acc += x[((b * 3 + ic) * 32 + iy) * 32 + ix] *
                           w[((oc * 3 + ic) * 3 + ky) * 3 + kx];
                }
            }
        m = fmaxf(m, fmaxf(acc * sc + sh, 0.f));
    }
    h1[t] = m;
}

// ---------------------------------------------------------------------------
// Kernel 4: conv2 implicit GEMM via WMMA. One block = one (b,y) row of 16
// x-positions (nt), 8 waves = 8 oc tiles sharing one LDS-staged B operand.
// LDS layout sH[yy][xx][ic], ic-pitch 72 halves: 144B xx-stride = 16B-aligned
// ds_load_b128 + bank-conflict-free (36 dwords; 36*l mod 64 distinct).
// ---------------------------------------------------------------------------
#define ICP 72

__global__ void __launch_bounds__(256)
conv2_wmma_kernel(const float* __restrict__ h1, const _Float16* __restrict__ packA,
                  const float* __restrict__ bias,
                  const float* __restrict__ g, const float* __restrict__ beta,
                  const float* __restrict__ mean, const float* __restrict__ var,
                  float* __restrict__ y2) {
    __shared__ _Float16 sH[3 * 18 * ICP];                // 7776 B
    int tid  = threadIdx.x;
    int lane = tid & 31, mt = tid >> 5;                  // wave = oc tile
    int half = lane >> 4, l15 = lane & 15;
    int nt = blockIdx.x;                                 // 512 blocks
    int s0 = nt * 16;
    int b = s0 >> 8, y = (s0 >> 4) & 15;                 // x = l15

    // stage h1[b, :, y-1..y+1, -1..16] with zero padding
    for (int idx = tid; idx < 3 * 18 * 64; idx += 256) {
        int ic = idx & 63;
        int xx = (idx >> 6) % 18;
        int yy = (idx >> 6) / 18;
        int iy = y + yy - 1, ix = xx - 1;
        float v = 0.f;
        if ((unsigned)iy < 16u && (unsigned)ix < 16u)
            v = h1[((b * C1 + ic) * 16 + iy) * 16 + ix];
        sH[(yy * 18 + xx) * ICP + ic] = (_Float16)v;
    }
    __syncthreads();

    const _Float16* aptr = packA + (size_t)mt * 18 * 512 + lane * 16;
    v8f c = {};
    #pragma unroll
    for (int kt = 0; kt < 18; ++kt) {
        int kk = kt >> 1, ich = kt & 1;
        int ky = kk / 3, kx = kk - ky * 3;
        v16h af = *(const v16h*)(aptr + kt * 512);       // 32B contiguous, coalesced
        const _Float16* bp = &sH[(ky * 18 + l15 + kx) * ICP + ich * 32 + half * 8];
        v8h b0 = *(const v8h*)bp;                        // ds_load_b128
        v8h b1 = *(const v8h*)(bp + 16);                 // ds_load_b128
        v16h bf = __builtin_shufflevector(b0, b1,
                    0, 1, 2, 3, 4, 5, 6, 7, 8, 9, 10, 11, 12, 13, 14, 15);
        c = __builtin_amdgcn_wmma_f32_16x16x32_f16(false, af, false, bf,
                                                   (short)0, c, false, false);
    }
    // C/D layout: VGPR r -> M = r + 8*half, N = l15
    #pragma unroll
    for (int r = 0; r < 8; ++r) {
        int oc = mt * 16 + r + (half << 3);
        float sc = g[oc] * rsqrtf(var[oc] + BN_EPS);
        float sh = beta[oc] - mean[oc] * sc;
        float v = fmaxf((c[r] + bias[oc]) * sc + sh, 0.f);
        y2[((b * C2 + oc) * 16 + y) * 16 + l15] = v;
    }
}

// ---------------------------------------------------------------------------
// Kernel 5: 2x2 maxpool -> x_feat [32,128,8,8] written directly to output
// ---------------------------------------------------------------------------
__global__ void pool2_kernel(const float* __restrict__ y2, float* __restrict__ xfeat) {
    int t = blockIdx.x * blockDim.x + threadIdx.x;       // 262144
    if (t >= BATCH * C2 * 8 * 8) return;
    int px = t & 7, py = (t >> 3) & 7, c = (t >> 6) & 127, b = t >> 13;
    const float* base = y2 + ((b * C2 + c) * 16 + py * 2) * 16 + px * 2;
    xfeat[t] = fmaxf(fmaxf(base[0], base[1]), fmaxf(base[16], base[17]));
}

// ---------------------------------------------------------------------------
// Kernel 6: L1-kNN. One wave per neighbor, 8 per block. Queries in LDS
// (pitch 1028: 16B-aligned float4 ds ops, 2-way conflict max), neighbor
// values broadcast via shuffles; lane = query index.
// ---------------------------------------------------------------------------
#define DCH    1024
#define FPITCH 1028

__global__ void __launch_bounds__(256)
knn_kernel(const float* __restrict__ xfeat, const float* __restrict__ nbr,
           const int* __restrict__ labels, float* __restrict__ ws) {
    extern __shared__ float smem[];
    float* sF  = smem;                        // 32 * FPITCH
    float* sNr = smem + 32 * FPITCH;          // 256
    float* sDr = sNr + 256;                   // 256
    int tid  = threadIdx.x;
    int lane = tid & 31;
    int wave = tid >> 5;
    int n = blockIdx.x * 8 + wave;            // 2500*8 = 20000

    float acc = 0.f;
    for (int ch = 0; ch < DFEAT / DCH; ++ch) {
        for (int idx = tid; idx < 32 * (DCH / 4); idx += 256) {
            int row = idx >> 8;               // DCH/4 == 256
            int c4  = idx & 255;
            float4 v = *(const float4*)(xfeat + row * DFEAT + ch * DCH + c4 * 4);
            *(float4*)(&sF[row * FPITCH + c4 * 4]) = v;   // ds_store_b128
        }
        __syncthreads();

        const float* nrow = nbr + (size_t)n * DFEAT + ch * DCH;
        const float* frow = &sF[lane * FPITCH];
        for (int dbase = 0; dbase < DCH; dbase += 32) {
            float v = nrow[dbase + lane];     // coalesced 32-float load
            __builtin_prefetch(nrow + dbase + 256, 0, 1);
            #pragma unroll
            for (int q = 0; q < 8; ++q) {
                float4 fq = *(const float4*)(frow + dbase + q * 4);  // ds_load_b128
                acc += fabsf(fq.x - __shfl(v, q * 4 + 0, 32));
                acc += fabsf(fq.y - __shfl(v, q * 4 + 1, 32));
                acc += fabsf(fq.z - __shfl(v, q * 4 + 2, 32));
                acc += fabsf(fq.w - __shfl(v, q * 4 + 3, 32));
            }
        }
        __syncthreads();
    }

    float wgt = __expf(-acc * (1.0f / 1000.0f));
    int pred = ((const int*)(ws + WS_PRED))[0];
    float wm = (labels[n] == pred) ? wgt : 0.f;
    sNr[tid] = wm;
    sDr[tid] = wgt;
    __syncthreads();
    if (wave == 0) {
        float a = 0.f, d = 0.f;
        #pragma unroll
        for (int k = 0; k < 8; ++k) { a += sNr[k * 32 + lane]; d += sDr[k * 32 + lane]; }
        atomicAdd(&ws[WS_NR + lane], a);
        atomicAdd(&ws[WS_DR + lane], d);
    }
}

// ---------------------------------------------------------------------------
// Kernel 7: out[b] = nr[b] / dr[b]
// ---------------------------------------------------------------------------
__global__ void finalize_kernel(const float* __restrict__ ws, float* __restrict__ out) {
    int b = threadIdx.x;
    if (b < 32) out[b] = ws[WS_NR + b] / ws[WS_DR + b];
}

// ---------------------------------------------------------------------------
extern "C" void kernel_launch(void* const* d_in, const int* in_sizes, int n_in,
                              void* d_out, int out_size, void* d_ws, size_t ws_size,
                              hipStream_t stream) {
    const float* x    = (const float*)d_in[0];
    const float* c1w  = (const float*)d_in[1];
    const float* c1b  = (const float*)d_in[2];
    const float* bn1g = (const float*)d_in[3];
    const float* bn1b = (const float*)d_in[4];
    const float* bn1m = (const float*)d_in[5];
    const float* bn1v = (const float*)d_in[6];
    const float* c2w  = (const float*)d_in[7];
    const float* c2b  = (const float*)d_in[8];
    const float* bn2g = (const float*)d_in[9];
    const float* bn2b = (const float*)d_in[10];
    const float* bn2m = (const float*)d_in[11];
    const float* bn2v = (const float*)d_in[12];
    const float* nbr  = (const float*)d_in[13];
    const int* labels = (const int*)d_in[14];

    float* out   = (float*)d_out;
    float* ws    = (float*)d_ws;
    float* xfeat = out + 32;                  // tuple: [32] nr/dr then [32,128,8,8]
    _Float16* packA = (_Float16*)(ws + WS_PACKA);

    init_stats_kernel<<<1, 256, 0, stream>>>(labels, ws);
    pack_w2_kernel<<<288, 256, 0, stream>>>(c2w, packA);
    conv1_kernel<<<2048, 256, 0, stream>>>(x, c1w, c1b, bn1g, bn1b, bn1m, bn1v, ws + WS_H1);
    conv2_wmma_kernel<<<512, 256, 0, stream>>>(ws + WS_H1, packA, c2b, bn2g, bn2b, bn2m, bn2v,
                                               ws + WS_Y2);
    pool2_kernel<<<1024, 256, 0, stream>>>(ws + WS_Y2, xfeat);
    size_t smem = (size_t)(32 * FPITCH + 512) * sizeof(float);   // ~133 KB (<320 KB WGP LDS)
    knn_kernel<<<2500, 256, smem, stream>>>(xfeat, nbr, labels, ws);
    finalize_kernel<<<1, 32, 0, stream>>>(ws, out);
}
